// LossFunction_16836271800471
// MI455X (gfx1250) — matrix operand order repaired and checked by reading, not compiled, not run on hardware
//
#include <hip/hip_runtime.h>
#include <hip/hip_bf16.h>
#include <math.h>

#define N_ROWS 4096
#define K_DIM  1024
#define EPSF   1e-8f

typedef __attribute__((ext_vector_type(16))) _Float16 v16h;
typedef __attribute__((ext_vector_type(8)))  _Float16 v8h;
typedef __attribute__((ext_vector_type(4)))  _Float16 v4h;
typedef __attribute__((ext_vector_type(8)))  float    v8f;

// Monotonic float -> u32 key (order-preserving over all finite floats)
__device__ __forceinline__ unsigned keyOf(float f) {
    unsigned b = __float_as_uint(f);
    return (b & 0x80000000u) ? ~b : (b | 0x80000000u);
}

// ---------------- fragment builders (ISA 7.12.2 layouts) ----------------
// A (16x32, MxK): lane l<16 -> row M=l, K={k..k+7, k+16..k+23}; lane>=16 -> K shifted by 8.
__device__ __forceinline__ v16h load_a16(const _Float16* __restrict__ row, int k, int hb) {
    v8h lo = *(const v8h*)(row + k + hb * 8);
    v8h hi = *(const v8h*)(row + k + 16 + hb * 8);
    v16h a;
    #pragma unroll
    for (int t = 0; t < 8; ++t) { a[t] = lo[t]; a[8 + t] = hi[t]; }
    return a;
}
// B (32x16, KxN): lane l<16 -> col N=l, K={k..k+7, k+16..k+23}; lane>=16 -> K shifted by 8.
__device__ __forceinline__ v16h load_b16(const _Float16* __restrict__ row, int k, int hb) {
    v8h lo = *(const v8h*)(row + k + hb * 8);
    v8h hi = *(const v8h*)(row + k + 16 + hb * 8);
    v16h b;
    #pragma unroll
    for (int t = 0; t < 8; ++t) { b[t] = lo[t]; b[8 + t] = hi[t]; }
    return b;
}
// f32 variants (fallback path, convert on the fly)
__device__ __forceinline__ v16h load_a32(const float* __restrict__ row, int k, int hb) {
    const float4* p = (const float4*)(row + k + hb * 8);
    const float4* q = (const float4*)(row + k + 16 + hb * 8);
    float4 f0 = p[0], f1 = p[1], f2 = q[0], f3 = q[1];
    v16h a;
    a[0]=(_Float16)f0.x; a[1]=(_Float16)f0.y; a[2]=(_Float16)f0.z; a[3]=(_Float16)f0.w;
    a[4]=(_Float16)f1.x; a[5]=(_Float16)f1.y; a[6]=(_Float16)f1.z; a[7]=(_Float16)f1.w;
    a[8]=(_Float16)f2.x; a[9]=(_Float16)f2.y; a[10]=(_Float16)f2.z; a[11]=(_Float16)f2.w;
    a[12]=(_Float16)f3.x; a[13]=(_Float16)f3.y; a[14]=(_Float16)f3.z; a[15]=(_Float16)f3.w;
    return a;
}
__device__ __forceinline__ v16h load_b32(const float* __restrict__ row, int k, int hb) {
    const float4* p = (const float4*)(row + k + hb * 8);
    const float4* q = (const float4*)(row + k + 16 + hb * 8);
    float4 f0 = p[0], f1 = p[1], f2 = q[0], f3 = q[1];
    v16h b;
    b[0]=(_Float16)f0.x; b[1]=(_Float16)f0.y; b[2]=(_Float16)f0.z; b[3]=(_Float16)f0.w;
    b[4]=(_Float16)f1.x; b[5]=(_Float16)f1.y; b[6]=(_Float16)f1.z; b[7]=(_Float16)f1.w;
    b[8]=(_Float16)f2.x; b[9]=(_Float16)f2.y; b[10]=(_Float16)f2.z; b[11]=(_Float16)f2.w;
    b[12]=(_Float16)f3.x; b[13]=(_Float16)f3.y; b[14]=(_Float16)f3.z; b[15]=(_Float16)f3.w;
    return b;
}

// argmax candidate: pack(sim, j) and merge across half-wave + global atomic
__device__ __forceinline__ void merge_row(unsigned long long p, int l15,
                                          unsigned long long* __restrict__ slot) {
    #pragma unroll
    for (int m = 1; m < 16; m <<= 1) {
        unsigned lo = __shfl_xor((unsigned)p, m, 32);
        unsigned hi = __shfl_xor((unsigned)(p >> 32), m, 32);
        unsigned long long q = ((unsigned long long)hi << 32) | lo;
        if (q > p) p = q;
    }
    if (l15 == 0) atomicMax(slot, p);
}

// ---------------- Kernel 1a (fast path): f32->f16 pack + norms + init ----------------
__global__ __launch_bounds__(256) void pack_norms_kernel(const float* __restrict__ x,
                                                         _Float16* __restrict__ pos16,
                                                         _Float16* __restrict__ anc16,
                                                         float* __restrict__ pn,
                                                         float* __restrict__ an,
                                                         unsigned long long* __restrict__ packed) {
    int i = blockIdx.x;
    int t = threadIdx.x;                       // 256 threads * float4 = 1024 elems
    const float4* pos = (const float4*)(x + (size_t)i * (2 * K_DIM));
    const float4* anc = (const float4*)(x + (size_t)i * (2 * K_DIM) + K_DIM);
    float4 p4 = pos[t];
    float4 a4 = anc[t];
    float s0 = p4.x*p4.x + p4.y*p4.y + p4.z*p4.z + p4.w*p4.w;
    float s1 = a4.x*a4.x + a4.y*a4.y + a4.z*a4.z + a4.w*a4.w;
    v4h ph; ph[0]=(_Float16)p4.x; ph[1]=(_Float16)p4.y; ph[2]=(_Float16)p4.z; ph[3]=(_Float16)p4.w;
    v4h ah; ah[0]=(_Float16)a4.x; ah[1]=(_Float16)a4.y; ah[2]=(_Float16)a4.z; ah[3]=(_Float16)a4.w;
    ((v4h*)(pos16 + (size_t)i * K_DIM))[t] = ph;
    ((v4h*)(anc16 + (size_t)i * K_DIM))[t] = ah;

    __shared__ float sh0[256], sh1[256];
    sh0[t] = s0; sh1[t] = s1;
    __syncthreads();
    for (int s = 128; s > 0; s >>= 1) {
        if (t < s) { sh0[t] += sh0[t + s]; sh1[t] += sh1[t + s]; }
        __syncthreads();
    }
    if (t == 0) {
        pn[i] = sqrtf(sh0[0]);
        an[i] = sqrtf(sh1[0]);
        packed[i] = 0ULL;
    }
}

// ---------------- Kernel 1b (fallback): norms + init only ----------------
__global__ __launch_bounds__(256) void norms_kernel(const float* __restrict__ x,
                                                    float* __restrict__ pn,
                                                    float* __restrict__ an,
                                                    unsigned long long* __restrict__ packed) {
    int i = blockIdx.x;
    const float* pos = x + (size_t)i * (2 * K_DIM);
    const float* anc = pos + K_DIM;
    int t = threadIdx.x;
    float s0 = 0.f, s1 = 0.f;
    for (int k = t; k < K_DIM; k += 256) {
        float a = pos[k]; s0 += a * a;
        float b = anc[k]; s1 += b * b;
    }
    __shared__ float sh0[256], sh1[256];
    sh0[t] = s0; sh1[t] = s1;
    __syncthreads();
    for (int s = 128; s > 0; s >>= 1) {
        if (t < s) { sh0[t] += sh0[t + s]; sh1[t] += sh1[t + s]; }
        __syncthreads();
    }
    if (t == 0) { pn[i] = sqrtf(sh0[0]); an[i] = sqrtf(sh1[0]); packed[i] = 0ULL; }
}

// ---------------- Kernel 2a (fast path): f16 WMMA GEMM, wave tile 32x64 ----------------
// Block = 8 waves (2 M x 4 N); block tile 64x256; 8 WMMAs / 12 b128 loads per K-step.
__global__ __launch_bounds__(256) void sim_argmax16_kernel(const _Float16* __restrict__ pos16,
                                                           const _Float16* __restrict__ anc16,
                                                           const float* __restrict__ pn,
                                                           const float* __restrict__ an,
                                                           unsigned long long* __restrict__ packed) {
    const int lane  = threadIdx.x & 31;
    const int wave  = threadIdx.x >> 5;
    const int waveM = wave & 1;
    const int waveN = wave >> 1;                 // 0..3
    const int mBase = blockIdx.y * 64  + waveM * 32;
    const int nBase = blockIdx.x * 256 + waveN * 64;
    const int l15   = lane & 15;
    const int hb    = lane >> 4;

    const _Float16* A0 = pos16 + (size_t)(mBase + l15)      * K_DIM;
    const _Float16* A1 = pos16 + (size_t)(mBase + 16 + l15) * K_DIM;
    const _Float16* B0 = anc16 + (size_t)(nBase + l15)      * K_DIM;
    const _Float16* B1 = anc16 + (size_t)(nBase + 16 + l15) * K_DIM;
    const _Float16* B2 = anc16 + (size_t)(nBase + 32 + l15) * K_DIM;
    const _Float16* B3 = anc16 + (size_t)(nBase + 48 + l15) * K_DIM;

    v8f a00={}, a01={}, a02={}, a03={};
    v8f a10={}, a11={}, a12={}, a13={};

    for (int k = 0; k < K_DIM; k += 32) {
        v16h fa0 = load_a16(A0, k, hb);
        v16h fa1 = load_a16(A1, k, hb);
        v16h fb0 = load_b16(B0, k, hb);
        v16h fb1 = load_b16(B1, k, hb);
        v16h fb2 = load_b16(B2, k, hb);
        v16h fb3 = load_b16(B3, k, hb);
        a00 = __builtin_amdgcn_wmma_f32_16x16x32_f16(false, fa0, false, fb0, (short)0, a00, false, false);
        a01 = __builtin_amdgcn_wmma_f32_16x16x32_f16(false, fa0, false, fb1, (short)0, a01, false, false);
        a02 = __builtin_amdgcn_wmma_f32_16x16x32_f16(false, fa0, false, fb2, (short)0, a02, false, false);
        a03 = __builtin_amdgcn_wmma_f32_16x16x32_f16(false, fa0, false, fb3, (short)0, a03, false, false);
        a10 = __builtin_amdgcn_wmma_f32_16x16x32_f16(false, fa1, false, fb0, (short)0, a10, false, false);
        a11 = __builtin_amdgcn_wmma_f32_16x16x32_f16(false, fa1, false, fb1, (short)0, a11, false, false);
        a12 = __builtin_amdgcn_wmma_f32_16x16x32_f16(false, fa1, false, fb2, (short)0, a12, false, false);
        a13 = __builtin_amdgcn_wmma_f32_16x16x32_f16(false, fa1, false, fb3, (short)0, a13, false, false);
    }

    float    anv[4];
    unsigned jinv[4];
    #pragma unroll
    for (int nt = 0; nt < 4; ++nt) {
        int j = nBase + nt * 16 + l15;
        anv[nt]  = an[j];
        jinv[nt] = ~(unsigned)j;                 // ties resolve to smallest j
    }

    #pragma unroll
    for (int mt = 0; mt < 2; ++mt) {
        v8f c0 = mt ? a10 : a00;
        v8f c1 = mt ? a11 : a01;
        v8f c2 = mt ? a12 : a02;
        v8f c3 = mt ? a13 : a03;
        #pragma unroll
        for (int r = 0; r < 8; ++r) {
            int   i   = mBase + mt * 16 + r + hb * 8;   // C/D layout: lanes>=16 -> M=r+8
            float pnr = pn[i];
            float s0 = c0[r] / fmaxf(pnr * anv[0], EPSF);
            float s1 = c1[r] / fmaxf(pnr * anv[1], EPSF);
            float s2 = c2[r] / fmaxf(pnr * anv[2], EPSF);
            float s3 = c3[r] / fmaxf(pnr * anv[3], EPSF);
            unsigned long long p0 = ((unsigned long long)keyOf(s0) << 32) | jinv[0];
            unsigned long long p1 = ((unsigned long long)keyOf(s1) << 32) | jinv[1];
            unsigned long long p2 = ((unsigned long long)keyOf(s2) << 32) | jinv[2];
            unsigned long long p3 = ((unsigned long long)keyOf(s3) << 32) | jinv[3];
            unsigned long long p  = p0 > p1 ? p0 : p1;
            if (p2 > p) p = p2;
            if (p3 > p) p = p3;
            merge_row(p, l15, packed + i);
        }
    }
}

// ---------------- Kernel 2b (fallback): f32 input, convert in-loop, wave tile 32x32 ----
__global__ __launch_bounds__(256) void sim_argmax32_kernel(const float* __restrict__ x,
                                                           const float* __restrict__ pn,
                                                           const float* __restrict__ an,
                                                           unsigned long long* __restrict__ packed) {
    const int lane  = threadIdx.x & 31;
    const int wave  = threadIdx.x >> 5;
    const int waveM = wave & 1;
    const int waveN = wave >> 1;
    const int mBase = blockIdx.y * 64  + waveM * 32;
    const int nBase = blockIdx.x * 128 + waveN * 32;
    const int l15   = lane & 15;
    const int hb    = lane >> 4;

    const float* A0 = x + (size_t)(mBase + l15)      * (2 * K_DIM);
    const float* A1 = x + (size_t)(mBase + 16 + l15) * (2 * K_DIM);
    const float* B0 = x + (size_t)(nBase + l15)      * (2 * K_DIM) + K_DIM;
    const float* B1 = x + (size_t)(nBase + 16 + l15) * (2 * K_DIM) + K_DIM;

    v8f acc00 = {}, acc01 = {}, acc10 = {}, acc11 = {};
    for (int k = 0; k < K_DIM; k += 32) {
        v16h fa0 = load_a32(A0, k, hb);
        v16h fa1 = load_a32(A1, k, hb);
        v16h fb0 = load_b32(B0, k, hb);
        v16h fb1 = load_b32(B1, k, hb);
        acc00 = __builtin_amdgcn_wmma_f32_16x16x32_f16(false, fa0, false, fb0, (short)0, acc00, false, false);
        acc01 = __builtin_amdgcn_wmma_f32_16x16x32_f16(false, fa0, false, fb1, (short)0, acc01, false, false);
        acc10 = __builtin_amdgcn_wmma_f32_16x16x32_f16(false, fa1, false, fb0, (short)0, acc10, false, false);
        acc11 = __builtin_amdgcn_wmma_f32_16x16x32_f16(false, fa1, false, fb1, (short)0, acc11, false, false);
    }

    const float    an0   = an[nBase + l15];
    const float    an1   = an[nBase + 16 + l15];
    const unsigned j0inv = ~(unsigned)(nBase + l15);
    const unsigned j1inv = ~(unsigned)(nBase + 16 + l15);

    #pragma unroll
    for (int mt = 0; mt < 2; ++mt) {
        v8f cN0 = mt ? acc10 : acc00;
        v8f cN1 = mt ? acc11 : acc01;
        #pragma unroll
        for (int r = 0; r < 8; ++r) {
            int   i   = mBase + mt * 16 + r + hb * 8;
            float pnr = pn[i];
            float s0  = cN0[r] / fmaxf(pnr * an0, EPSF);
            float s1  = cN1[r] / fmaxf(pnr * an1, EPSF);
            unsigned long long p0 = ((unsigned long long)keyOf(s0) << 32) | j0inv;
            unsigned long long p1 = ((unsigned long long)keyOf(s1) << 32) | j1inv;
            unsigned long long p  = p0 > p1 ? p0 : p1;
            merge_row(p, l15, packed + i);
        }
    }
}

// ---------------- Kernel 3: outputs ----------------
__global__ __launch_bounds__(256) void finalize_kernel(const unsigned long long* __restrict__ packed,
                                                       float* __restrict__ out) {
    __shared__ int sh[256];
    int t = threadIdx.x;
    int cnt = 0;
    for (int i = t; i < N_ROWS; i += 256) {
        unsigned j = ~(unsigned)(packed[i] & 0xFFFFFFFFull);
        cnt += (j == (unsigned)i);
    }
    sh[t] = cnt;
    __syncthreads();
    for (int s = 128; s > 0; s >>= 1) {
        if (t < s) sh[t] += sh[t + s];
        __syncthreads();
    }
    if (t == 0) {
        out[0] = 1.0f;                                   // nloss forward == 1.0 exactly
        out[1] = 100.0f * (float)sh[0] / (float)N_ROWS;  // prec1
    }
}

extern "C" void kernel_launch(void* const* d_in, const int* in_sizes, int n_in,
                              void* d_out, int out_size, void* d_ws, size_t ws_size,
                              hipStream_t stream) {
    (void)in_sizes; (void)n_in; (void)out_size;
    const float* x   = (const float*)d_in[0];
    float*       out = (float*)d_out;

    // ws layout: [0,32K) packed u64[4096] | [32K,48K) pn | [48K,64K) an | [64K,...) f16 operands
    unsigned long long* packed = (unsigned long long*)d_ws;
    float* pn = (float*)((char*)d_ws + 32768);
    float* an = (float*)((char*)d_ws + 49152);

    const size_t halfBytes = (size_t)N_ROWS * K_DIM * 2;       // 8 MB per operand
    const size_t need      = 65536 + 2 * halfBytes;            // ~16.06 MB

    if (ws_size >= need) {
        _Float16* pos16 = (_Float16*)((char*)d_ws + 65536);
        _Float16* anc16 = (_Float16*)((char*)d_ws + 65536 + halfBytes);
        pack_norms_kernel<<<N_ROWS, 256, 0, stream>>>(x, pos16, anc16, pn, an, packed);
        dim3 grid(N_ROWS / 256, N_ROWS / 64);                  // 16 x 64 blocks
        sim_argmax16_kernel<<<grid, 256, 0, stream>>>(pos16, anc16, pn, an, packed);
    } else {
        norms_kernel<<<N_ROWS, 256, 0, stream>>>(x, pn, an, packed);
        dim3 grid(N_ROWS / 128, N_ROWS / 64);                  // 32 x 64 blocks
        sim_argmax32_kernel<<<grid, 256, 0, stream>>>(x, pn, an, packed);
    }
    finalize_kernel<<<1, 256, 0, stream>>>(packed, out);
}
